// LocalRNN_52132313039487
// MI455X (gfx1250) — matrix operand order repaired and checked by reading, not compiled, not run on hardware
//
#include <hip/hip_runtime.h>

typedef __attribute__((ext_vector_type(16))) _Float16 v16h;
typedef __attribute__((ext_vector_type(2)))  _Float16 h2;
typedef __attribute__((ext_vector_type(8)))  float    v8f;

#define DMODEL 512
#define NGATE  2048
#define NFRAME 8192
#define KWIN   9
#define GPW    2          // 16-window groups per workgroup
#define MROWS  (16 * GPW) // 32 windows per workgroup
#define BM     4          // GEMM: M-tiles per wave
#define BN     2          // GEMM: N-tiles per wave

__device__ __forceinline__ float sigmoidf_(float x) { return 1.0f / (1.0f + __expf(-x)); }
// branchless tanh via hardware v_exp/v_rcp; exact at +/-inf saturation
__device__ __forceinline__ float tanhf_(float x) { return 1.0f - 2.0f / (__expf(2.0f * x) + 1.0f); }

// ---- WMMA fragment loaders (CDNA5 16x16x32 f16 layouts, wave32) ----

// A (16x32, row-major source, leading dim lda). p points at element (row0, k0).
// lane 0-15: M=lane, K pairs {2v,2v+1} (+16 for v>=4); lanes 16-31: same +8.
// Compiler vectorizes this into two b128 loads per lane.
__device__ __forceinline__ v16h load_a_frag(const _Float16* p, int lda, int lane) {
  const _Float16* r = p + (size_t)(lane & 15) * lda;
  const int hi = (lane >> 4) << 3;
  v16h a;
#pragma unroll
  for (int v = 0; v < 8; ++v) {
    const int kk = ((v & 3) << 1) + ((v >> 2) << 4) + hi;
    h2 pr = *(const h2*)(r + kk);
    a[2 * v]     = pr.x;
    a[2 * v + 1] = pr.y;
  }
  return a;
}

// B (32x16): weight pre-transposed to K x N row-major; wt points at (k0, n0).
// lane supplies N column (lane&15); elements walk K in the same order as A.
__device__ __forceinline__ v16h load_b_frag(const _Float16* wt, int ldn, int lane) {
  const _Float16* c = wt + (lane & 15);
  const int hi = (lane >> 4) << 3;
  v16h b;
#pragma unroll
  for (int s = 0; s < 16; ++s) {
    const int kk = (s & 7) + ((s >> 3) << 4) + hi;
    b[s] = c[(size_t)kk * ldn];
  }
  return b;
}

// ---- prep kernels ----

// out[k*N + n] = (f16) in[n*K + k]   (in: N rows x K cols, f32)
__global__ void k_cvt_transpose(const float* __restrict__ in, _Float16* __restrict__ out,
                                int N, int K) {
  int i = blockIdx.x * blockDim.x + threadIdx.x;
  if (i >= N * K) return;
  int k = i / N, n = i - k * N;
  out[i] = (_Float16)in[(size_t)n * K + k];
}

__global__ void k_bias_sum(const float* __restrict__ a, const float* __restrict__ b,
                           float* __restrict__ o, int n) {
  int i = blockIdx.x * blockDim.x + threadIdx.x;
  if (i < n) o[i] = a[i] + b[i];
}

// GLU: u[row, col] = a * sigmoid(b), a = x[row, col], b = x[row, 256+col], f16 out
__global__ void k_glu(const float* __restrict__ x, _Float16* __restrict__ u) {
  int i = blockIdx.x * blockDim.x + threadIdx.x; // over 8192*256
  int row = i >> 8, col = i & 255;
  float a = x[(size_t)row * 512 + col];
  float b = x[(size_t)row * 512 + 256 + col];
  u[i] = (_Float16)(a * sigmoidf_(b));
}

// ---- f16 WMMA GEMM, 64x32 per wave: C[M,N] = f16(A[M,K] @ Wt[K,N] + bias[n]) ----
__global__ void k_gemm_f16(const _Float16* __restrict__ A, const _Float16* __restrict__ Wt,
                           const float* __restrict__ bias, _Float16* __restrict__ C,
                           int M, int N, int K) {
  const int lane = threadIdx.x & 31;
  const int wave = blockIdx.x * (blockDim.x >> 5) + (threadIdx.x >> 5);
  const int nblk = N >> 5;                 // 32 columns per wave
  const int mb = wave / nblk;
  const int nb = wave - mb * nblk;
  if (mb >= (M >> 6)) return;              // 64 rows per wave
  const int row0 = mb << 6, col0 = nb << 5;

  v8f zero8 = {};
  v8f acc[BM][BN];
#pragma unroll
  for (int i = 0; i < BM; ++i)
#pragma unroll
    for (int j = 0; j < BN; ++j) acc[i][j] = zero8;

  for (int k0 = 0; k0 < K; k0 += 32) {
    v16h a[BM];
#pragma unroll
    for (int i = 0; i < BM; ++i)
      a[i] = load_a_frag(A + (size_t)(row0 + 16 * i) * K + k0, K, lane);
    v16h b[BN];
#pragma unroll
    for (int j = 0; j < BN; ++j)
      b[j] = load_b_frag(Wt + (size_t)k0 * N + col0 + 16 * j, N, lane);
#pragma unroll
    for (int i = 0; i < BM; ++i)
#pragma unroll
      for (int j = 0; j < BN; ++j)
        acc[i][j] = __builtin_amdgcn_wmma_f32_16x16x32_f16(false, a[i], false, b[j],
                                                           (short)0, acc[i][j],
                                                           false, false);
  }

  const int nlo = lane & 15;
  const int mhi = (lane >> 4) << 3;
#pragma unroll
  for (int j = 0; j < BN; ++j) {
    const int n = col0 + 16 * j + nlo;
    const float bn = bias[n];
#pragma unroll
    for (int i = 0; i < BM; ++i) {
      const int mbase = row0 + 16 * i + mhi;
#pragma unroll
      for (int v = 0; v < 8; ++v)
        C[(size_t)(mbase + v) * N + n] = (_Float16)(acc[i][j][v] + bn);
    }
  }
}

// ---- local LSTM recurrence: 32 windows per WG, 9 steps, gates via WMMA.
// Per step, the 128 KB gate-x slab is DMA'd into LDS with
// global_load_async_to_lds_b128, overlapped with the WMMA phase. ----
__global__ __launch_bounds__(512)
void k_lstm(const _Float16* __restrict__ gxh, const float* __restrict__ bsum,
            const _Float16* __restrict__ whh_t, float* __restrict__ out) {
  __shared__ _Float16 hbuf[MROWS][DMODEL]; // 32 KB f16 hidden state
  __shared__ _Float16 slab[MROWS][NGATE];  // 128 KB gate-x staging (total 160 KB)
  const int tid  = threadIdx.x;
  const int lane = tid & 31;
  const int wv   = tid >> 5;               // 16 waves
  const int wbase = blockIdx.x * MROWS;    // first window index of block
  const int d0   = wv << 5;                // this wave's 32-wide hidden chunk
  const int nlo  = lane & 15;
  const int mhi  = (lane >> 4) << 3;

  for (int i = tid; i < MROWS * DMODEL; i += 512) ((_Float16*)hbuf)[i] = (_Float16)0.0f;

  v8f zero8 = {};
  v8f c[GPW][2];
#pragma unroll
  for (int mt = 0; mt < GPW; ++mt)
#pragma unroll
    for (int s = 0; s < 2; ++s) c[mt][s] = zero8;

  const uint32_t slab_lds = (uint32_t)(size_t)(&slab[0][0]); // low 32b = LDS byte addr

  __syncthreads();

  for (int t = 0; t < KWIN; ++t) {
    // ---- issue async DMA of this step's gate-x slab into LDS ----
    // slab rows m=0..31 <- gxh row (wbase+m) + t - 8 (clamped; padded rows are
    // ignored at use and replaced by bsum). 8192 16B chunks over 512 threads.
#pragma unroll
    for (int i = 0; i < 16; ++i) {
      const int chunk = i * 512 + tid;        // 256 chunks per 4 KB row
      const int m = chunk >> 8;
      const int boff = (chunk & 255) << 4;
      int grow = wbase + m + t - 8;
      grow = grow < 0 ? 0 : grow;
      const uint32_t la = slab_lds + (uint32_t)(m << 12) + (uint32_t)boff;
      const uint64_t ga = (uint64_t)(size_t)((const char*)gxh + ((size_t)grow << 12) + boff);
      asm volatile("global_load_async_to_lds_b128 %0, %1, off"
                   :: "v"(la), "v"(ga) : "memory");
    }

    // ---- gates = h @ w_hh^T  (f16 WMMA, f32 accumulate, C starts at 0) ----
    v8f acc[GPW][4][2];
#pragma unroll
    for (int mt = 0; mt < GPW; ++mt)
#pragma unroll
      for (int q = 0; q < 4; ++q)
#pragma unroll
        for (int s = 0; s < 2; ++s) acc[mt][q][s] = zero8;

    for (int k0 = 0; k0 < DMODEL; k0 += 32) {
      v16h a0 = load_a_frag(&hbuf[0][k0],  DMODEL, lane);
      v16h a1 = load_a_frag(&hbuf[16][k0], DMODEL, lane);
#pragma unroll
      for (int q = 0; q < 4; ++q)
#pragma unroll
        for (int s = 0; s < 2; ++s) {
          const int n0 = q * DMODEL + d0 + (s << 4);
          v16h b = load_b_frag(whh_t + (size_t)k0 * NGATE + n0, NGATE, lane);
          acc[0][q][s] = __builtin_amdgcn_wmma_f32_16x16x32_f16(
              false, a0, false, b, (short)0, acc[0][q][s], false, false);
          acc[1][q][s] = __builtin_amdgcn_wmma_f32_16x16x32_f16(
              false, a1, false, b, (short)0, acc[1][q][s], false, false);
        }
    }

    // wait for this wave's DMA, then barrier: makes all waves' slab writes
    // visible AND retires all hbuf reads before the rewrite below.
    asm volatile("s_wait_asynccnt 0" ::: "memory");
    __syncthreads();

    // ---- elementwise LSTM cell update; write new h (f16) to LDS ----
#pragma unroll
    for (int mt = 0; mt < GPW; ++mt)
#pragma unroll
      for (int s = 0; s < 2; ++s)
#pragma unroll
        for (int v = 0; v < 8; ++v) {
          const int m = (mt << 4) + mhi + v;
          const int w = wbase + m;
          const int l = w & 511;
          const bool pad = (l + t < 8);
          const int nc = d0 + (s << 4) + nlo;
          float gi = acc[mt][0][s][v] + (pad ? bsum[nc]          : (float)slab[m][nc]);
          float gf = acc[mt][1][s][v] + (pad ? bsum[512 + nc]    : (float)slab[m][512 + nc]);
          float gg = acc[mt][2][s][v] + (pad ? bsum[1024 + nc]   : (float)slab[m][1024 + nc]);
          float go = acc[mt][3][s][v] + (pad ? bsum[1536 + nc]   : (float)slab[m][1536 + nc]);
          const float cv = sigmoidf_(gf) * c[mt][s][v] + sigmoidf_(gi) * tanhf_(gg);
          const float hv = sigmoidf_(go) * tanhf_(cv);
          c[mt][s][v] = cv;
          hbuf[m][nc] = (_Float16)hv;
          if (t == KWIN - 1) out[(size_t)w * DMODEL + nc] = hv;
        }
    __syncthreads(); // slab reads done before next step's DMA rewrites it
  }
}

// ---- host launcher ----
extern "C" void kernel_launch(void* const* d_in, const int* in_sizes, int n_in,
                              void* d_out, int out_size, void* d_ws, size_t ws_size,
                              hipStream_t stream) {
  const float* x      = (const float*)d_in[0]; // (16,512,512)
  const float* conv_w = (const float*)d_in[1]; // (512,256)
  const float* conv_b = (const float*)d_in[2]; // (512)
  const float* w_ih   = (const float*)d_in[3]; // (2048,512)
  const float* w_hh   = (const float*)d_in[4]; // (2048,512)
  const float* b_ih   = (const float*)d_in[5]; // (2048)
  const float* b_hh   = (const float*)d_in[6]; // (2048)
  float* out = (float*)d_out;                  // (16,512,512) f32

  char* ws = (char*)d_ws;
  _Float16* gxh     = (_Float16*)(ws);              // 8192*2048*2 = 32 MB
  _Float16* proj_h  = (_Float16*)(ws + 33554432);   // 8192*512*2  =  8 MB
  _Float16* u_h     = (_Float16*)(ws + 41943040);   // 8192*256*2  =  4 MB
  _Float16* convw_t = (_Float16*)(ws + 46137344);   // 256*512*2   = 256 KB
  _Float16* wih_t   = (_Float16*)(ws + 46399488);   // 512*2048*2  =  2 MB
  _Float16* whh_t   = (_Float16*)(ws + 48496640);   // 512*2048*2  =  2 MB
  float*    bsum    = (float*)(ws + 50593792);      // 2048*4      =  8 KB
  // total workspace: 50,601,984 bytes (~48.3 MB)

  // weight prep: f32 -> f16, transposed to K x N for contiguous B fragments
  k_cvt_transpose<<<(512 * 256 + 255) / 256, 256, 0, stream>>>(conv_w, convw_t, 512, 256);
  k_cvt_transpose<<<(2048 * 512 + 255) / 256, 256, 0, stream>>>(w_ih, wih_t, 2048, 512);
  k_cvt_transpose<<<(2048 * 512 + 255) / 256, 256, 0, stream>>>(w_hh, whh_t, 2048, 512);
  k_bias_sum<<<(2048 + 255) / 256, 256, 0, stream>>>(b_ih, b_hh, bsum, 2048);

  // GLU -> u (8192 x 256 f16)
  k_glu<<<8192, 256, 0, stream>>>(x, u_h);

  // proj = u @ conv_w^T + conv_b  (M=8192,N=512,K=256), 64x32 per wave
  k_gemm_f16<<<256, 256, 0, stream>>>(u_h, convw_t, conv_b, proj_h, 8192, 512, 256);
  // GX = proj @ w_ih^T + (b_ih+b_hh)  (M=8192,N=2048,K=512) -> f16
  k_gemm_f16<<<1024, 256, 0, stream>>>(proj_h, wih_t, bsum, gxh, 8192, 2048, 512);

  // 9-step local LSTM recurrence, 32 windows per workgroup
  k_lstm<<<NFRAME / MROWS, 512, 0, stream>>>(gxh, bsum, whh_t, out);
}
